// TimeAwareMultiHeadAttention_39994735460876
// MI455X (gfx1250) — compile-verified, matrix operand order checked by reading, not run on hardware
//
#include <hip/hip_runtime.h>

// Problem sizes (compile-time)
#define B_   4
#define L_   1024
#define HID_ 512
#define NH_  8
#define DH_  64
#define BL_  (B_ * L_)       // 4096
#define LL_  (L_ * L_)       // 1048576

typedef __attribute__((ext_vector_type(16))) __bf16 v16bf;
typedef __attribute__((ext_vector_type(8)))  float  v8f;
typedef unsigned int  v4u __attribute__((ext_vector_type(4)));
typedef unsigned int  v8u __attribute__((ext_vector_type(8)));
typedef unsigned int  v2u __attribute__((ext_vector_type(2)));
typedef unsigned short u16;

// ---------------------------------------------------------------- helpers

__device__ __forceinline__ u16 f2bf(float f) {
  union { float f; unsigned int u; } x; x.f = f;
  unsigned int u = x.u;
  u += 0x7FFFu + ((u >> 16) & 1u);      // round-to-nearest-even
  return (u16)(u >> 16);
}
__device__ __forceinline__ float bf2f(u16 h) {
  union { unsigned int u; float f; } x; x.u = ((unsigned int)h) << 16;
  return x.f;
}

// 16-bit A/B fragment for V_WMMA_*_16X16X32: lanes 0-15 hold row M=lane with
// K {0..7, 16..23}; lanes 16-31 hold row M=lane-16 with K {8..15, 24..31}.
// Two 16B loads per lane from row-major [16 x ld] data (global or LDS).
__device__ __forceinline__ v16bf load_frag(const u16* __restrict__ base, int ld, int lane) {
  const int r  = lane & 15;
  const int kh = (lane >> 4) << 3;              // 0 or 8
  const u16* p = base + (size_t)r * ld + kh;    // 16B aligned by construction
  union { v4u q[2]; v16bf v; } f;
  f.q[0] = *(const v4u*)(p);                    // K = kh .. kh+7
  f.q[1] = *(const v4u*)(p + 16);               // K = 16+kh .. 16+kh+7
  return f.v;
}

#define WMMA_BF16(a, b, c) \
  __builtin_amdgcn_wmma_f32_16x16x32_bf16(false, (a), false, (b), (short)0, (c), false, false)

// C/D layout: VGPR r -> M = r (lanes 0-15) or M = 8+r (lanes 16-31); N = lane%16.
#define C_ROW(lane) (((lane) >> 4) << 3)
#define C_COL(lane) ((lane) & 15)

// ----- CDNA5 async global->LDS copy (ASYNCcnt) : 16 bytes per call ---------
__device__ __forceinline__ void async_copy_b128(unsigned lds_addr, const void* gaddr) {
  asm volatile("global_load_async_to_lds_b128 %0, %1, off"
               :: "v"(lds_addr), "v"((unsigned long long)(size_t)gaddr)
               : "memory");
}
__device__ __forceinline__ void wait_asynccnt0() {
  asm volatile("s_wait_asynccnt 0x0" ::: "memory");
}

// ----- CDNA5 Tensor Data Mover: 2D bf16 tile (rows x 32) -> LDS ------------
// D# group0: count=1, lds_addr, global_addr, type=2 ("image")
// D# group1: data_size=1 (2B), tensor dims/strides, tile 32 x rows
__device__ __forceinline__ void tdm_load_tile(const u16* gsrc, unsigned lds_addr,
                                              unsigned rows, unsigned tensor_d0,
                                              unsigned tensor_d1, unsigned stride0) {
  unsigned long long ga = (unsigned long long)(size_t)gsrc;
  v4u g0;
  g0.x = 1u;                                                  // count=1, user mode
  g0.y = lds_addr;                                            // LDS byte address
  g0.z = (unsigned)(ga & 0xFFFFFFFFu);                        // global_addr[31:0]
  g0.w = ((unsigned)(ga >> 32) & 0x01FFFFFFu) | 0x80000000u;  // addr[56:32] | type=2
  v8u g1;
  g1[0] = 0x00010000u;                                        // data_size=1 -> 2 bytes
  g1[1] = (tensor_d0 & 0xFFFFu) << 16;                        // tensor_dim0[15:0]
  g1[2] = (tensor_d0 >> 16) | ((tensor_d1 & 0xFFFFu) << 16);  // dim0 hi | dim1 lo
  g1[3] = (tensor_d1 >> 16) | (32u << 16);                    // dim1 hi | tile_dim0=32
  g1[4] = rows & 0xFFFFu;                                     // tile_dim1 (tile_dim2=0)
  g1[5] = stride0;                                            // tensor_dim0_stride lo32
  g1[6] = 0u;                                                 // stride0 hi | stride1 lo
  g1[7] = 0u;
  asm volatile("tensor_load_to_lds %0, %1" :: "s"(g0), "s"(g1) : "memory");
}

// ---------------------------------------------------------------- kernels

// fp32 -> bf16 pack
__global__ void k_cvt(const float* __restrict__ in, u16* __restrict__ out, int n) {
  int i = blockIdx.x * blockDim.x + threadIdx.x;
  if (i < n) out[i] = f2bf(in[i]);
}

// Y[M,N] = A[M,K] @ W[N,K]^T + bias (+ residual X when RESID) -- LDS-staged,
// double-buffered: A tile via global_load_async_to_lds_b128 (ASYNCcnt),
// W tile via the Tensor Data Mover (TENSORcnt), fragments via ds_load_b128.
// Block = 256 threads = 8 waves; block tile 128(M) x 64(N); K step 32.
template <bool RESID>
__global__ void __launch_bounds__(256)
k_gemm_lds(const u16* __restrict__ A, const u16* __restrict__ W,
           const float* __restrict__ bias, const float* __restrict__ X,
           u16* __restrict__ Ybf, float* __restrict__ Yf,
           int M, int N, int K) {
  __shared__ __attribute__((aligned(16))) u16 As[2][128][32];  // 2 x 8 KB
  __shared__ __attribute__((aligned(16))) u16 Ws[2][64][32];   // 2 x 4 KB

  const int tid  = threadIdx.x;
  const int lane = tid & 31;
  const int wv   = tid >> 5;
  const int nb   = N >> 6;
  const int bm   = blockIdx.x / nb;
  const int bn   = blockIdx.x % nb;
  const int m0   = bm << 7;        // 128-row block
  const int n0   = bn << 6;        // 64-col block
  const int nsteps = K >> 5;

  const u16* Abase = A + (size_t)m0 * K;
  const u16* Wbase = W + (size_t)n0 * K;

  // prologue: fill buffer 0
  {
    const int ch0 = tid << 1;                   // 512 chunks of 16B for A tile
#pragma unroll
    for (int e = 0; e < 2; ++e) {
      const int ch = ch0 + e;
      const int row = ch >> 2, cc = (ch & 3) << 3;
      async_copy_b128((unsigned)(size_t)&As[0][row][cc],
                      Abase + (size_t)row * K + cc);
    }
    if (wv == 0)
      tdm_load_tile(Wbase, (unsigned)(size_t)&Ws[0][0][0], 64, K, N, K);
  }

  v8f acc[4] = {};
  for (int i = 0; i < nsteps; ++i) {
    const int cur = i & 1;
    wait_asynccnt0();                           // per-wave: own A-tile copies
    __builtin_amdgcn_s_wait_tensorcnt(0);       // wave 0: TDM W tile done
    __syncthreads();                            // LDS tile visible to all waves

    if (i + 1 < nsteps) {                       // issue next tile into other buf
      const int kb = (i + 1) << 5;
      const int ch0 = tid << 1;
#pragma unroll
      for (int e = 0; e < 2; ++e) {
        const int ch = ch0 + e;
        const int row = ch >> 2, cc = (ch & 3) << 3;
        async_copy_b128((unsigned)(size_t)&As[cur ^ 1][row][cc],
                        Abase + (size_t)row * K + kb + cc);
      }
      if (wv == 0)
        tdm_load_tile(Wbase + kb, (unsigned)(size_t)&Ws[cur ^ 1][0][0], 64, K, N, K);
    }

    // compute current buffer: 1 A-frag + 4 W-frags -> 4 WMMAs
    v16bf a = load_frag(&As[cur][wv << 4][0], 32, lane);
#pragma unroll
    for (int t = 0; t < 4; ++t) {
      v16bf b = load_frag(&Ws[cur][t << 4][0], 32, lane);
      acc[t] = WMMA_BF16(a, b, acc[t]);
    }
    __syncthreads();                            // done reading before next overwrite
  }

  const int crow = C_ROW(lane), ccol = C_COL(lane);
  const int mrow = m0 + (wv << 4) + crow;
#pragma unroll
  for (int t = 0; t < 4; ++t) {
    const int col = n0 + (t << 4) + ccol;
    const float bs = bias[col];
#pragma unroll
    for (int r = 0; r < 8; ++r) {
      const size_t row = (size_t)(mrow + r);
      if (RESID)
        Yf[row * N + col] = acc[t][r] + bs + X[row * N + col];
      else
        Ybf[row * N + col] = f2bf(acc[t][r] + bs);
    }
  }
}

// time gate: per batch  s = tq @ x^T  (K = 512), epilogue computes
// sigmoid(tow1*tanh(log1p(|ti-tj|)*tw1+tib) + tow2*tanh(s) + tob)  -> f32
__global__ void __launch_bounds__(256)
k_gate(const u16* __restrict__ TQ, const u16* __restrict__ X,
       const float* __restrict__ tseq,
       const float* __restrict__ tw1,  const float* __restrict__ tib,
       const float* __restrict__ tow1, const float* __restrict__ tow2,
       const float* __restrict__ tob,  float* __restrict__ tatt) {
  const int lane = threadIdx.x & 31;
  const int w    = (blockIdx.x * blockDim.x + threadIdx.x) >> 5;
  const int tilesM = L_ >> 4, tilesN = L_ >> 6;       // 64, 16
  const int b   = w / (tilesM * tilesN);
  const int rem = w % (tilesM * tilesN);
  const int mt = rem / tilesN, ng = rem % tilesN;
  const int i0 = mt << 4, j0 = ng << 6;
  const u16* Ab = TQ + (size_t)(b * L_ + i0) * HID_;
  const u16* Bb = X  + (size_t)(b * L_ + j0) * HID_;
  v8f acc[4] = {};
  for (int kb = 0; kb < HID_; kb += 32) {
    __builtin_prefetch(Ab + kb + 128, 0, 0);
    v16bf a = load_frag(Ab + kb, HID_, lane);
#pragma unroll
    for (int t = 0; t < 4; ++t) {
      v16bf bb = load_frag(Bb + (size_t)(t * 16) * HID_ + kb, HID_, lane);
      acc[t] = WMMA_BF16(a, bb, acc[t]);
    }
  }
  const int crow = C_ROW(lane), ccol = C_COL(lane);
#pragma unroll
  for (int t = 0; t < 4; ++t) {
    const int j = j0 + t * 16 + ccol;
    const float tj = tseq[b * L_ + j];
#pragma unroll
    for (int r = 0; r < 8; ++r) {
      const int i = i0 + crow + r;
      const float ti = tseq[b * L_ + i];
      const int ij = i * L_ + j;
      const float decay = tanhf(log1pf(fabsf(ti - tj)) * tw1[ij] + tib[ij]);
      const float g = tow1[ij] * decay + tow2[ij] * tanhf(acc[t][r]) + tob[ij];
      tatt[(size_t)b * LL_ + ij] = 1.0f / (1.0f + __expf(-g));
    }
  }
}

// scores = (q @ k^T) * gate * 1/sqrt(DH) + mask  -> bf16 [BH, L, L]
__global__ void __launch_bounds__(256)
k_scores(const u16* __restrict__ Q, const u16* __restrict__ Kb,
         const float* __restrict__ tatt, const float* __restrict__ mask,
         u16* __restrict__ sc) {
  const int lane = threadIdx.x & 31;
  const int w    = (blockIdx.x * blockDim.x + threadIdx.x) >> 5;
  const int tilesM = 64, tilesN = 16;
  const int bh  = w / (tilesM * tilesN);
  const int rem = w % (tilesM * tilesN);
  const int mt = rem / tilesN, ng = rem % tilesN;
  const int b = bh >> 3, h = bh & 7;
  const int i0 = mt << 4, j0 = ng << 6;
  const u16* Ab = Q  + (size_t)(b * L_ + i0) * HID_ + h * DH_;
  const u16* Bb = Kb + (size_t)(b * L_ + j0) * HID_ + h * DH_;
  v8f acc[4] = {};
  for (int kb = 0; kb < DH_; kb += 32) {
    v16bf a = load_frag(Ab + kb, HID_, lane);
#pragma unroll
    for (int t = 0; t < 4; ++t) {
      v16bf bb = load_frag(Bb + (size_t)(t * 16) * HID_ + kb, HID_, lane);
      acc[t] = WMMA_BF16(a, bb, acc[t]);
    }
  }
  const int gb = (b * NH_ + h) % B_;   // faithful tile(NH,1,1)+reshape scramble
  const int crow = C_ROW(lane), ccol = C_COL(lane);
#pragma unroll
  for (int t = 0; t < 4; ++t) {
    const int j = j0 + t * 16 + ccol;
#pragma unroll
    for (int r = 0; r < 8; ++r) {
      const int i = i0 + crow + r;
      const size_t ij = (size_t)i * L_ + j;
      const float v = acc[t][r] * tatt[(size_t)gb * LL_ + ij] * 0.125f
                    + mask[(size_t)b * LL_ + ij];
      sc[((size_t)bh * L_ + i) * L_ + j] = f2bf(v);
    }
  }
}

// in-place row softmax over 1024 bf16 elements (one block per row)
__global__ void __launch_bounds__(256) k_softmax(u16* __restrict__ sc) {
  __shared__ float red[256];
  const int t = threadIdx.x;
  u16* p = sc + (size_t)blockIdx.x * L_ + t * 4;
  v2u d = *(const v2u*)p;
  float v0 = bf2f((u16)(d.x & 0xFFFF)), v1 = bf2f((u16)(d.x >> 16));
  float v2 = bf2f((u16)(d.y & 0xFFFF)), v3 = bf2f((u16)(d.y >> 16));
  red[t] = fmaxf(fmaxf(v0, v1), fmaxf(v2, v3));
  __syncthreads();
  for (int s = 128; s > 0; s >>= 1) { if (t < s) red[t] = fmaxf(red[t], red[t + s]); __syncthreads(); }
  const float m = red[0];
  __syncthreads();
  float e0 = __expf(v0 - m), e1 = __expf(v1 - m), e2 = __expf(v2 - m), e3 = __expf(v3 - m);
  red[t] = e0 + e1 + e2 + e3;
  __syncthreads();
  for (int s = 128; s > 0; s >>= 1) { if (t < s) red[t] += red[t + s]; __syncthreads(); }
  const float inv = 1.0f / red[0];
  v2u o;
  o.x = (unsigned)f2bf(e0 * inv) | ((unsigned)f2bf(e1 * inv) << 16);
  o.y = (unsigned)f2bf(e2 * inv) | ((unsigned)f2bf(e3 * inv) << 16);
  *(v2u*)p = o;
}

// v [B,L,H,DH] (bf16) -> vT [B,H,DH,L] (bf16) so probs@v gets contiguous B-frags
__global__ void k_vT(const u16* __restrict__ V, u16* __restrict__ VT) {
  int idx = blockIdx.x * blockDim.x + threadIdx.x;
  if (idx >= BL_ * HID_) return;
  const int c  = idx & (HID_ - 1);
  const int bl = idx >> 9;            // / HID_
  const int j  = bl & (L_ - 1);
  const int b  = bl >> 10;
  const int h  = c >> 6;              // / DH_
  const int d  = c & 63;
  VT[((size_t)(b * NH_ + h) * DH_ + d) * L_ + j] = V[idx];
}

// ctx = probs @ v : per (b,h) 1024x64 output, K = L = 1024
__global__ void __launch_bounds__(256)
k_ctx(const u16* __restrict__ P, const u16* __restrict__ VT, u16* __restrict__ C) {
  const int lane = threadIdx.x & 31;
  const int w    = (blockIdx.x * blockDim.x + threadIdx.x) >> 5;
  const int bh = w >> 6, mt = w & 63;
  const int b = bh >> 3, h = bh & 7;
  const int i0 = mt << 4;
  const u16* Ab = P  + ((size_t)bh * L_ + i0) * L_;
  const u16* Bb = VT + (size_t)bh * DH_ * L_;
  v8f acc[4] = {};
  for (int kb = 0; kb < L_; kb += 32) {
    __builtin_prefetch(Ab + kb + 128, 0, 0);
    v16bf a = load_frag(Ab + kb, L_, lane);
#pragma unroll
    for (int t = 0; t < 4; ++t) {
      v16bf bb = load_frag(Bb + (size_t)(t * 16) * L_ + kb, L_, lane);
      acc[t] = WMMA_BF16(a, bb, acc[t]);
    }
  }
  const int crow = C_ROW(lane), ccol = C_COL(lane);
#pragma unroll
  for (int t = 0; t < 4; ++t) {
    const int d = t * 16 + ccol;
#pragma unroll
    for (int r = 0; r < 8; ++r)
      C[(size_t)(b * L_ + i0 + crow + r) * HID_ + h * DH_ + d] = f2bf(acc[t][r]);
  }
}

// LayerNorm over last dim (512), one block (256 thr, 2 elems/thr) per row
__global__ void __launch_bounds__(256)
k_ln(const float* __restrict__ H, const float* __restrict__ g,
     const float* __restrict__ be, float* __restrict__ out) {
  __shared__ float s1[256], s2[256];
  const int t = threadIdx.x;
  const float* p = H + (size_t)blockIdx.x * HID_;
  const float a = p[t], b = p[t + 256];
  s1[t] = a + b;
  s2[t] = a * a + b * b;
  __syncthreads();
  for (int s = 128; s > 0; s >>= 1) {
    if (t < s) { s1[t] += s1[t + s]; s2[t] += s2[t + s]; }
    __syncthreads();
  }
  const float mu  = s1[0] * (1.0f / HID_);
  const float var = s2[0] * (1.0f / HID_) - mu * mu;
  const float inv = rsqrtf(var + 1e-12f);
  float* o = out + (size_t)blockIdx.x * HID_;
  o[t]       = (a - mu) * inv * g[t]       + be[t];
  o[t + 256] = (b - mu) * inv * g[t + 256] + be[t + 256];
}

// ---------------------------------------------------------------- launch

extern "C" void kernel_launch(void* const* d_in, const int* in_sizes, int n_in,
                              void* d_out, int out_size, void* d_ws, size_t ws_size,
                              hipStream_t stream) {
  (void)in_sizes; (void)n_in; (void)out_size; (void)ws_size;
  const float* x    = (const float*)d_in[0];
  const float* tseq = (const float*)d_in[1];
  const float* mask = (const float*)d_in[2];
  const float* Wq   = (const float*)d_in[3];
  const float* bq   = (const float*)d_in[4];
  const float* Wk   = (const float*)d_in[5];
  const float* bk   = (const float*)d_in[6];
  const float* Wv   = (const float*)d_in[7];
  const float* bv   = (const float*)d_in[8];
  const float* Wtq  = (const float*)d_in[9];
  const float* btq  = (const float*)d_in[10];
  const float* tw1  = (const float*)d_in[11];
  const float* tib  = (const float*)d_in[12];
  const float* tow1 = (const float*)d_in[13];
  const float* tow2 = (const float*)d_in[14];
  const float* tob  = (const float*)d_in[15];
  const float* Wd   = (const float*)d_in[16];
  const float* bd   = (const float*)d_in[17];
  const float* lng  = (const float*)d_in[18];
  const float* lnb  = (const float*)d_in[19];
  float* out = (float*)d_out;

  char* ws = (char*)d_ws;
  auto alloc = [&](size_t bytes) {
    char* p = ws;
    ws += (bytes + 255) & ~(size_t)255;
    return p;
  };
  u16* xb   = (u16*)alloc((size_t)BL_ * HID_ * 2);
  u16* wqb  = (u16*)alloc((size_t)HID_ * HID_ * 2);
  u16* wkb  = (u16*)alloc((size_t)HID_ * HID_ * 2);
  u16* wvb  = (u16*)alloc((size_t)HID_ * HID_ * 2);
  u16* wtqb = (u16*)alloc((size_t)HID_ * HID_ * 2);
  u16* wdb  = (u16*)alloc((size_t)HID_ * HID_ * 2);
  u16* qb   = (u16*)alloc((size_t)BL_ * HID_ * 2);
  u16* kbuf = (u16*)alloc((size_t)BL_ * HID_ * 2);
  u16* vb   = (u16*)alloc((size_t)BL_ * HID_ * 2);
  u16* tqb  = (u16*)alloc((size_t)BL_ * HID_ * 2);
  float* tatt = (float*)alloc((size_t)B_ * LL_ * 4);
  u16* sc   = (u16*)alloc((size_t)B_ * NH_ * LL_ * 2);
  u16* vT   = (u16*)alloc((size_t)BL_ * HID_ * 2);
  u16* ctxb = (u16*)alloc((size_t)BL_ * HID_ * 2);
  float* hb = (float*)alloc((size_t)BL_ * HID_ * 4);

  const int NT = 256;
  // pack fp32 -> bf16
  k_cvt<<<(BL_ * HID_ + NT - 1) / NT, NT, 0, stream>>>(x, xb, BL_ * HID_);
  k_cvt<<<(HID_ * HID_ + NT - 1) / NT, NT, 0, stream>>>(Wq,  wqb,  HID_ * HID_);
  k_cvt<<<(HID_ * HID_ + NT - 1) / NT, NT, 0, stream>>>(Wk,  wkb,  HID_ * HID_);
  k_cvt<<<(HID_ * HID_ + NT - 1) / NT, NT, 0, stream>>>(Wv,  wvb,  HID_ * HID_);
  k_cvt<<<(HID_ * HID_ + NT - 1) / NT, NT, 0, stream>>>(Wtq, wtqb, HID_ * HID_);
  k_cvt<<<(HID_ * HID_ + NT - 1) / NT, NT, 0, stream>>>(Wd,  wdb,  HID_ * HID_);

  // projections: 128x64 block tiles -> (4096/128)*(512/64) = 256 blocks
  k_gemm_lds<false><<<256, NT, 0, stream>>>(xb, wqb,  bq,  nullptr, qb,   nullptr, BL_, HID_, HID_);
  k_gemm_lds<false><<<256, NT, 0, stream>>>(xb, wkb,  bk,  nullptr, kbuf, nullptr, BL_, HID_, HID_);
  k_gemm_lds<false><<<256, NT, 0, stream>>>(xb, wvb,  bv,  nullptr, vb,   nullptr, BL_, HID_, HID_);
  k_gemm_lds<false><<<256, NT, 0, stream>>>(xb, wtqb, btq, nullptr, tqb,  nullptr, BL_, HID_, HID_);

  // time gate: 4 * 64 * 16 = 4096 waves = 512 blocks
  k_gate<<<512, NT, 0, stream>>>(tqb, xb, tseq, tw1, tib, tow1, tow2, tob, tatt);

  // scores: 32 * 64 * 16 = 32768 waves = 4096 blocks
  k_scores<<<4096, NT, 0, stream>>>(qb, kbuf, tatt, mask, sc);

  // softmax: one block per (b,h,i) row
  k_softmax<<<B_ * NH_ * L_, NT, 0, stream>>>(sc);

  // v transpose + ctx GEMM: 32 * 64 = 2048 waves = 256 blocks
  k_vT<<<(BL_ * HID_ + NT - 1) / NT, NT, 0, stream>>>(vb, vT);
  k_ctx<<<256, NT, 0, stream>>>(sc, vT, ctxb);

  // output projection (+bias+residual, fp32) and LayerNorm
  k_gemm_lds<true><<<256, NT, 0, stream>>>(ctxb, wdb, bd, x, nullptr, hb, BL_, HID_, HID_);
  k_ln<<<BL_, NT, 0, stream>>>(hb, lng, lnb, out);
}
